// CTRNN_45896020525280
// MI455X (gfx1250) — matrix operand order repaired and checked by reading, not compiled, not run on hardware
//
#include <hip/hip_runtime.h>

// ---------------------------------------------------------------------------
// CTRNN on gfx1250 (MI455X): xproj GEMM + persistent recurrent GEMM on
// v_wmma_f32_16x16x32_bf16 with split-bf16 (truncated-hi + RNE-lo) for
// ~fp32 accuracy. W_in is pre-split once into L2-resident bf16 planes;
// the recurrent kernel keeps W_h in LDS and exchanges h across 16
// persistent WGs through a bf16 ping-pong buffer with release/acquire
// counters (h slices double as the mandatory f32 output stores).
// ---------------------------------------------------------------------------

typedef __attribute__((ext_vector_type(16))) __bf16 v16bf;
typedef __attribute__((ext_vector_type(8)))  float  v8f;

#define T_STEPS 2048
#define B_SZ    64
#define I_SZ    128
#define H_SZ    512
#define SB      520   // padded LDS row stride (bf16 elems); 1040B rows, 16B-aligned

static __device__ __forceinline__ unsigned bcu(float f) {
  return __builtin_bit_cast(unsigned, f);
}
static __device__ __forceinline__ float bcf(unsigned u) {
  return __builtin_bit_cast(float, u);
}
static __device__ __forceinline__ unsigned short bfbits(float f) {
  return __builtin_bit_cast(unsigned short, (__bf16)f);   // native cvt, RNE
}
static __device__ __forceinline__ unsigned pack2(unsigned short a, unsigned short b) {
  return (unsigned)a | ((unsigned)b << 16);
}
// Split float4 into packed truncated-hi bf16 pairs and RNE-lo bf16 pairs.
static __device__ __forceinline__ void split4(const float4 v, uint2& ph, uint2& pl) {
  unsigned u0 = bcu(v.x), u1 = bcu(v.y), u2 = bcu(v.z), u3 = bcu(v.w);
  ph.x = (u0 >> 16) | (u1 & 0xFFFF0000u);
  ph.y = (u2 >> 16) | (u3 & 0xFFFF0000u);
  pl.x = pack2(bfbits(v.x - bcf(u0 & 0xFFFF0000u)),
               bfbits(v.y - bcf(u1 & 0xFFFF0000u)));
  pl.y = pack2(bfbits(v.z - bcf(u2 & 0xFFFF0000u)),
               bfbits(v.w - bcf(u3 & 0xFFFF0000u)));
}

union FragAB { v16bf v; uint4 q[2]; unsigned d[8]; };
union FragC  { v8f   v; float f[8]; };

static __device__ __forceinline__ v8f wmma_bf16(v16bf a, v16bf b, v8f c) {
  return __builtin_amdgcn_wmma_f32_16x16x32_bf16(false, a, false, b,
                                                 (short)0, c, false, false);
}

// ---------------------------------------------------------------------------
// Prep: split W_in (512x128 f32) once into bf16 hi/lo planes in scratch.
// ---------------------------------------------------------------------------
extern "C" __global__ void __launch_bounds__(256)
ctrnn_prep_win(const float* __restrict__ W_in, unsigned short* __restrict__ wHi,
               unsigned short* __restrict__ wLo)
{
  const int c = blockIdx.x * 256 + threadIdx.x;        // chunk of 4 floats
  if (c >= (H_SZ * I_SZ) / 4) return;
  const float4 v = *(const float4*)(W_in + (long)c * 4);
  uint2 ph, pl; split4(v, ph, pl);
  *(uint2*)(wHi + (long)c * 4) = ph;
  *(uint2*)(wLo + (long)c * 4) = pl;
}

// ---------------------------------------------------------------------------
// Kernel 1 (fast): xproj[t,b,:] = x[t,b,:] @ W_in^T + b_in  (into d_out)
// B fragments come pre-split from scratch: no conversions in inner loop.
// Block = 256 thr (8 waves): 16 M-rows x 512 N-cols; wave w owns cols [64w,+64)
// ---------------------------------------------------------------------------
extern "C" __global__ void __launch_bounds__(256)
ctrnn_xproj_fast(const float* __restrict__ x, const unsigned short* __restrict__ wHi,
                 const unsigned short* __restrict__ wLo,
                 const float* __restrict__ b_in, float* __restrict__ out)
{
  const int lane = threadIdx.x & 31;
  const int wave = threadIdx.x >> 5;
  const long rowBase = (long)blockIdx.x * 16;
  const int  mrow = lane & 15;
  const int  aOff = (lane < 16) ? 0 : 8;    // A 16-bit layout: 2nd half-lanes K+8
  const int  bOff = (lane < 16) ? 0 : 8;    // B 16-bit layout: 2nd half-lanes K+8

  const float* xrow = x + (rowBase + mrow) * I_SZ;

  FragC C[4];
#pragma unroll
  for (int t = 0; t < 4; ++t)
#pragma unroll
    for (int j = 0; j < 8; ++j) C[t].f[j] = 0.0f;

#pragma unroll
  for (int kt = 0; kt < 4; ++kt) {
    const int k0 = kt * 32;
    float av[16];
    *(float4*)&av[0]  = *(const float4*)(xrow + k0 + aOff);
    *(float4*)&av[4]  = *(const float4*)(xrow + k0 + aOff + 4);
    *(float4*)&av[8]  = *(const float4*)(xrow + k0 + aOff + 16);
    *(float4*)&av[12] = *(const float4*)(xrow + k0 + aOff + 20);
    FragAB ahi, alo;
#pragma unroll
    for (int p = 0; p < 8; ++p) {
      const unsigned u0 = bcu(av[2 * p]), u1 = bcu(av[2 * p + 1]);
      ahi.d[p] = (u0 >> 16) | (u1 & 0xFFFF0000u);
      alo.d[p] = pack2(bfbits(av[2 * p]     - bcf(u0 & 0xFFFF0000u)),
                       bfbits(av[2 * p + 1] - bcf(u1 & 0xFFFF0000u)));
    }
#pragma unroll
    for (int t = 0; t < 4; ++t) {
      const int n = wave * 64 + t * 16 + (lane & 15);
      const unsigned short* wh = wHi + (long)n * I_SZ + k0 + bOff;
      const unsigned short* wl = wLo + (long)n * I_SZ + k0 + bOff;
      FragAB bhi, blo;
      bhi.q[0] = *(const uint4*)(wh);
      bhi.q[1] = *(const uint4*)(wh + 16);
      blo.q[0] = *(const uint4*)(wl);
      blo.q[1] = *(const uint4*)(wl + 16);
      C[t].v = wmma_bf16(ahi.v, bhi.v, C[t].v);
      C[t].v = wmma_bf16(ahi.v, blo.v, C[t].v);
      C[t].v = wmma_bf16(alo.v, bhi.v, C[t].v);
    }
  }

  const int mAdd = (lane >> 4) << 3;   // C layout: M = r + 8*(lane>=16)
#pragma unroll
  for (int t = 0; t < 4; ++t) {
    const int n = wave * 64 + t * 16 + (lane & 15);
    const float bias = b_in[n];
#pragma unroll
    for (int r = 0; r < 8; ++r) {
      const long row = rowBase + r + mAdd;
      out[row * H_SZ + n] = C[t].f[r] + bias;
    }
  }
}

// ---------------------------------------------------------------------------
// Kernel 1 (fallback, tiny workspace): converts W_in fragments in-loop.
// ---------------------------------------------------------------------------
extern "C" __global__ void __launch_bounds__(256)
ctrnn_xproj_conv(const float* __restrict__ x, const float* __restrict__ W_in,
                 const float* __restrict__ b_in, float* __restrict__ out)
{
  const int lane = threadIdx.x & 31;
  const int wave = threadIdx.x >> 5;
  const long rowBase = (long)blockIdx.x * 16;
  const int  mrow = lane & 15;
  const int  aOff = (lane < 16) ? 0 : 8;
  const int  bOff = (lane < 16) ? 0 : 8;

  const float* xrow = x + (rowBase + mrow) * I_SZ;

  FragC C[4];
#pragma unroll
  for (int t = 0; t < 4; ++t)
#pragma unroll
    for (int j = 0; j < 8; ++j) C[t].f[j] = 0.0f;

#pragma unroll
  for (int kt = 0; kt < 4; ++kt) {
    const int k0 = kt * 32;
    float av[16];
    *(float4*)&av[0]  = *(const float4*)(xrow + k0 + aOff);
    *(float4*)&av[4]  = *(const float4*)(xrow + k0 + aOff + 4);
    *(float4*)&av[8]  = *(const float4*)(xrow + k0 + aOff + 16);
    *(float4*)&av[12] = *(const float4*)(xrow + k0 + aOff + 20);
    FragAB ahi, alo;
#pragma unroll
    for (int p = 0; p < 8; ++p) {
      const unsigned u0 = bcu(av[2 * p]), u1 = bcu(av[2 * p + 1]);
      ahi.d[p] = (u0 >> 16) | (u1 & 0xFFFF0000u);
      alo.d[p] = pack2(bfbits(av[2 * p]     - bcf(u0 & 0xFFFF0000u)),
                       bfbits(av[2 * p + 1] - bcf(u1 & 0xFFFF0000u)));
    }
#pragma unroll
    for (int t = 0; t < 4; ++t) {
      const int n = wave * 64 + t * 16 + (lane & 15);
      const float* wrow = W_in + (long)n * I_SZ + k0 + bOff;
      float bv[16];
      *(float4*)&bv[0]  = *(const float4*)(wrow);
      *(float4*)&bv[4]  = *(const float4*)(wrow + 4);
      *(float4*)&bv[8]  = *(const float4*)(wrow + 16);
      *(float4*)&bv[12] = *(const float4*)(wrow + 20);
      FragAB bhi, blo;
#pragma unroll
      for (int p = 0; p < 8; ++p) {
        const unsigned u0 = bcu(bv[2 * p]), u1 = bcu(bv[2 * p + 1]);
        bhi.d[p] = (u0 >> 16) | (u1 & 0xFFFF0000u);
        blo.d[p] = pack2(bfbits(bv[2 * p]     - bcf(u0 & 0xFFFF0000u)),
                         bfbits(bv[2 * p + 1] - bcf(u1 & 0xFFFF0000u)));
      }
      C[t].v = wmma_bf16(ahi.v, bhi.v, C[t].v);
      C[t].v = wmma_bf16(ahi.v, blo.v, C[t].v);
      C[t].v = wmma_bf16(alo.v, bhi.v, C[t].v);
    }
  }

  const int mAdd = (lane >> 4) << 3;
#pragma unroll
  for (int t = 0; t < 4; ++t) {
    const int n = wave * 64 + t * 16 + (lane & 15);
    const float bias = b_in[n];
#pragma unroll
    for (int r = 0; r < 8; ++r) {
      const long row = rowBase + r + mAdd;
      out[row * H_SZ + n] = C[t].f[r] + bias;
    }
  }
}

// ---------------------------------------------------------------------------
// Kernel 2 (fast path): persistent recurrence with bf16 hi/lo ping-pong
// exchange buffer in global scratch (L2-resident). No per-step LDS staging.
//   exch layout: [slot(2)][plane(2)][B_SZ*H_SZ] bf16, row-major [b][h].
//   phase counters: cnt[phase*4 + mg]; phase p armed when h_{p-1} published.
// ---------------------------------------------------------------------------
extern "C" __global__ void __launch_bounds__(256)
ctrnn_recur_fast(const float* __restrict__ hidden, const float* __restrict__ W_h,
                 const float* __restrict__ b_h, float* __restrict__ out,
                 unsigned int* __restrict__ cnt, unsigned short* __restrict__ exch)
{
  extern __shared__ unsigned short smem[];
  unsigned short* Bhi = smem;                // [128][SB]
  unsigned short* Blo = Bhi + 128 * SB;      // [128][SB]

  const int wg   = blockIdx.x;
  const int mg   = wg >> 2;                  // batch group: rows 16*mg..+16
  const int ng   = wg & 3;                   // col group:   cols 128*ng..+128
  const int tid  = threadIdx.x;
  const int wave = tid >> 5;                 // wave owns N-tile `wave` (16 cols)
  const int lane = tid & 31;

  const long planeSz = (long)B_SZ * H_SZ;    // 32768 elems per exch plane

  // ---- one-time: stage W_h slice as bf16 hi/lo in LDS, [n_local][k] ----
  for (int c = tid; c < 128 * (H_SZ / 4); c += 256) {
    const int nl = c >> 7;
    const int kc = (c & 127) * 4;
    const float4 w = *(const float4*)(W_h + (long)(ng * 128 + nl) * H_SZ + kc);
    uint2 ph, pl; split4(w, ph, pl);
    *(uint2*)(Bhi + nl * SB + kc) = ph;
    *(uint2*)(Blo + nl * SB + kc) = pl;
  }

  // ---- one-time: publish hidden (h_{-1}) into exch slot 0 ----
  for (int c = tid; c < 16 * (I_SZ / 4); c += 256) {   // 16 rows x 128 own cols
    const int m  = c >> 5;
    const int kc = (c & 31) * 4;
    const int row = mg * 16 + m;
    const int col = ng * 128 + kc;
    const float4 v = *(const float4*)(hidden + (long)row * H_SZ + col);
    uint2 ph, pl; split4(v, ph, pl);
    *(uint2*)(exch + (long)row * H_SZ + col) = ph;
    *(uint2*)(exch + planeSz + (long)row * H_SZ + col) = pl;
  }
  __threadfence();
  __syncthreads();
  if (tid == 0) {
    __hip_atomic_fetch_add(&cnt[mg], 1u, __ATOMIC_RELEASE, __HIP_MEMORY_SCOPE_AGENT);
  }

  const int nLoc  = lane & 15;
  const int gnCol = ng * 128 + wave * 16 + nLoc;
  const int mAdd  = (lane >> 4) << 3;
  const int aOff  = (lane < 16) ? 0 : 8;
  const int bOff  = (lane < 16) ? 0 : 8;
  const float bh  = b_h[gnCol];

  float hprev[8];
#pragma unroll
  for (int r = 0; r < 8; ++r)
    hprev[r] = hidden[(long)(mg * 16 + r + mAdd) * H_SZ + gnCol];

  const unsigned short* bH = Bhi + (wave * 16 + nLoc) * SB;
  const unsigned short* bL = Blo + (wave * 16 + nLoc) * SB;
  const long sliceSz = (long)B_SZ * H_SZ;

  for (int t = 0; t < T_STEPS; ++t) {
    // ---- wait for phase t: h_{t-1} fully published ----
    if (tid == 0) {
      while (__hip_atomic_load(&cnt[t * 4 + mg], __ATOMIC_ACQUIRE,
                               __HIP_MEMORY_SCOPE_AGENT) < 4u) {
        __builtin_amdgcn_s_sleep(2);
      }
    }
    __syncthreads();
    __threadfence();                       // block-wide acquire visibility

    // ---- own xp slice (d_out still holds xproj at [t]) ----
    float xp[8];
#pragma unroll
    for (int r = 0; r < 8; ++r) {
      const long row = (long)t * B_SZ + mg * 16 + r + mAdd;
      xp[r] = out[row * H_SZ + gnCol];
    }

    // ---- C = h_{t-1} @ W_h^T for own 16x16 N-tile (K = 512) ----
    const unsigned short* eBase = exch + (long)(t & 1) * planeSz * 2;
    const unsigned short* aH = eBase + (long)(mg * 16 + nLoc) * H_SZ;
    const unsigned short* aL = eBase + planeSz + (long)(mg * 16 + nLoc) * H_SZ;

    FragC C;
#pragma unroll
    for (int j = 0; j < 8; ++j) C.f[j] = 0.0f;
#pragma unroll
    for (int kt = 0; kt < 16; ++kt) {
      const int k0 = kt * 32;
      FragAB ahi, alo, bhi, blo;
      ahi.q[0] = *(const uint4*)(aH + k0 + aOff);
      ahi.q[1] = *(const uint4*)(aH + k0 + aOff + 16);
      alo.q[0] = *(const uint4*)(aL + k0 + aOff);
      alo.q[1] = *(const uint4*)(aL + k0 + aOff + 16);
      bhi.q[0] = *(const uint4*)(bH + k0 + bOff);
      bhi.q[1] = *(const uint4*)(bH + k0 + bOff + 16);
      blo.q[0] = *(const uint4*)(bL + k0 + bOff);
      blo.q[1] = *(const uint4*)(bL + k0 + bOff + 16);
      C.v = wmma_bf16(ahi.v, bhi.v, C.v);
      C.v = wmma_bf16(ahi.v, blo.v, C.v);
      C.v = wmma_bf16(alo.v, bhi.v, C.v);
    }

    // ---- epilogue: relu + leaky update; publish f32 output + bf16 exchange ----
    unsigned short* oH = exch + (long)((t + 1) & 1) * planeSz * 2;
    unsigned short* oL = oH + planeSz;
#pragma unroll
    for (int r = 0; r < 8; ++r) {
      float pre = C.f[r] + xp[r] + bh;
      float hn  = pre > 0.0f ? pre : 0.0f;
      float h   = hprev[r] * 0.9f + hn * 0.1f;
      hprev[r]  = h;
      const int  brow = mg * 16 + r + mAdd;
      const long row  = (long)t * B_SZ + brow;
      out[row * H_SZ + gnCol] = h;
      const unsigned hu = bcu(h);
      oH[(long)brow * H_SZ + gnCol] = (unsigned short)(hu >> 16);
      oL[(long)brow * H_SZ + gnCol] = bfbits(h - bcf(hu & 0xFFFF0000u));
      if (t == T_STEPS - 1) {
        out[(long)T_STEPS * sliceSz + (long)brow * H_SZ + gnCol] = h;
      }
    }

    // ---- release: arm phase t+1 ----
    __threadfence();
    __syncthreads();
    if (tid == 0) {
      __hip_atomic_fetch_add(&cnt[(t + 1) * 4 + mg], 1u, __ATOMIC_RELEASE,
                             __HIP_MEMORY_SCOPE_AGENT);
    }
  }
}

// ---------------------------------------------------------------------------
// Kernel 2 (fallback, small workspace): per-step LDS staging of h_{t-1}.
// ---------------------------------------------------------------------------
extern "C" __global__ void __launch_bounds__(256)
ctrnn_recur_staged(const float* __restrict__ hidden, const float* __restrict__ W_h,
                   const float* __restrict__ b_h, float* __restrict__ out,
                   unsigned int* __restrict__ cnt)
{
  extern __shared__ unsigned short smem[];
  unsigned short* Bhi = smem;                // [128][SB]
  unsigned short* Blo = Bhi + 128 * SB;
  unsigned short* Ahi = Blo + 128 * SB;      // [16][SB]
  unsigned short* Alo = Ahi + 16 * SB;

  const int wg   = blockIdx.x;
  const int mg   = wg >> 2;
  const int ng   = wg & 3;
  const int tid  = threadIdx.x;
  const int wave = tid >> 5;
  const int lane = tid & 31;

  for (int c = tid; c < 128 * (H_SZ / 4); c += 256) {
    const int nl = c >> 7;
    const int kc = (c & 127) * 4;
    const float4 w = *(const float4*)(W_h + (long)(ng * 128 + nl) * H_SZ + kc);
    uint2 ph, pl; split4(w, ph, pl);
    *(uint2*)(Bhi + nl * SB + kc) = ph;
    *(uint2*)(Blo + nl * SB + kc) = pl;
  }

  const int nLoc  = lane & 15;
  const int gnCol = ng * 128 + wave * 16 + nLoc;
  const int mAdd  = (lane >> 4) << 3;
  const int aOff  = (lane < 16) ? 0 : 8;
  const int bOff  = (lane < 16) ? 0 : 8;
  const float bh  = b_h[gnCol];

  float hprev[8];
#pragma unroll
  for (int r = 0; r < 8; ++r)
    hprev[r] = hidden[(long)(mg * 16 + r + mAdd) * H_SZ + gnCol];

  const long sliceSz = (long)B_SZ * H_SZ;

  for (int t = 0; t < T_STEPS; ++t) {
    if (t > 0 && tid == 0) {
      while (__hip_atomic_load(&cnt[(t - 1) * 4 + mg], __ATOMIC_ACQUIRE,
                               __HIP_MEMORY_SCOPE_AGENT) < 4u) {
        __builtin_amdgcn_s_sleep(2);
      }
    }
    __syncthreads();
    if (t > 0) __threadfence();

    float xp[8];
#pragma unroll
    for (int r = 0; r < 8; ++r) {
      const long row = (long)t * B_SZ + mg * 16 + r + mAdd;
      xp[r] = out[row * H_SZ + gnCol];
    }

    const float* hsrc = (t == 0) ? (hidden + (long)mg * 16 * H_SZ)
                                 : (out + (long)(t - 1) * sliceSz + (long)mg * 16 * H_SZ);
    for (int c = tid; c < 16 * (H_SZ / 4); c += 256) {
      const int m  = c >> 7;
      const int kc = (c & 127) * 4;
      const float4 v = *(const float4*)(hsrc + (long)m * H_SZ + kc);
      uint2 ph, pl; split4(v, ph, pl);
      *(uint2*)(Ahi + m * SB + kc) = ph;
      *(uint2*)(Alo + m * SB + kc) = pl;
    }
    __syncthreads();

    FragC C;
#pragma unroll
    for (int j = 0; j < 8; ++j) C.f[j] = 0.0f;
    const unsigned short* aH = Ahi + nLoc * SB;
    const unsigned short* aL = Alo + nLoc * SB;
    const unsigned short* bH = Bhi + (wave * 16 + nLoc) * SB;
    const unsigned short* bL = Blo + (wave * 16 + nLoc) * SB;
#pragma unroll
    for (int kt = 0; kt < 16; ++kt) {
      const int k0 = kt * 32;
      FragAB ahi, alo, bhi, blo;
      ahi.q[0] = *(const uint4*)(aH + k0 + aOff);
      ahi.q[1] = *(const uint4*)(aH + k0 + aOff + 16);
      alo.q[0] = *(const uint4*)(aL + k0 + aOff);
      alo.q[1] = *(const uint4*)(aL + k0 + aOff + 16);
      bhi.q[0] = *(const uint4*)(bH + k0 + bOff);
      bhi.q[1] = *(const uint4*)(bH + k0 + bOff + 16);
      blo.q[0] = *(const uint4*)(bL + k0 + bOff);
      blo.q[1] = *(const uint4*)(bL + k0 + bOff + 16);
      C.v = wmma_bf16(ahi.v, bhi.v, C.v);
      C.v = wmma_bf16(ahi.v, blo.v, C.v);
      C.v = wmma_bf16(alo.v, bhi.v, C.v);
    }

#pragma unroll
    for (int r = 0; r < 8; ++r) {
      float pre = C.f[r] + xp[r] + bh;
      float hn  = pre > 0.0f ? pre : 0.0f;
      float h   = hprev[r] * 0.9f + hn * 0.1f;
      hprev[r]  = h;
      const long row = (long)t * B_SZ + mg * 16 + r + mAdd;
      out[row * H_SZ + gnCol] = h;
      if (t == T_STEPS - 1) {
        out[(long)T_STEPS * sliceSz + (long)(mg * 16 + r + mAdd) * H_SZ + gnCol] = h;
      }
    }

    __threadfence();
    __syncthreads();
    if (tid == 0) {
      __hip_atomic_fetch_add(&cnt[t * 4 + mg], 1u, __ATOMIC_RELEASE,
                             __HIP_MEMORY_SCOPE_AGENT);
    }
  }
}

// ---------------------------------------------------------------------------
extern "C" void kernel_launch(void* const* d_in, const int* in_sizes, int n_in,
                              void* d_out, int out_size, void* d_ws, size_t ws_size,
                              hipStream_t stream) {
  const float* x      = (const float*)d_in[0];
  const float* hidden = (const float*)d_in[1];
  const float* W_in   = (const float*)d_in[2];
  const float* b_in   = (const float*)d_in[3];
  const float* W_h    = (const float*)d_in[4];
  const float* b_h    = (const float*)d_in[5];
  float* out = (float*)d_out;

  // workspace layout
  unsigned int* cnt = (unsigned int*)d_ws;
  const size_t cntBytes  = (size_t)(T_STEPS + 1) * 4 * sizeof(unsigned int); // 32,784
  const size_t exchOff   = 33792;
  const size_t exchBytes = 4ul * B_SZ * H_SZ * sizeof(unsigned short);       // 262,144
  const size_t winOff    = exchOff + exchBytes;                              // 295,936
  const size_t winBytes  = 2ul * H_SZ * I_SZ * sizeof(unsigned short);       // 262,144

  // zero the per-phase arrival counters (capture-safe async memset)
  hipMemsetAsync(cnt, 0, cntBytes, stream);

  // 1) xproj into d_out[t,b,:]
  if (ws_size >= winOff + winBytes) {
    unsigned short* wHi = (unsigned short*)((char*)d_ws + winOff);
    unsigned short* wLo = wHi + H_SZ * I_SZ;
    ctrnn_prep_win<<<(H_SZ * I_SZ / 4 + 255) / 256, 256, 0, stream>>>(W_in, wHi, wLo);
    ctrnn_xproj_fast<<<(T_STEPS * B_SZ) / 16, 256, 0, stream>>>(x, wHi, wLo, b_in, out);
  } else {
    ctrnn_xproj_conv<<<(T_STEPS * B_SZ) / 16, 256, 0, stream>>>(x, W_in, b_in, out);
  }

  // 2) persistent recurrence (16 WGs)
  if (ws_size >= exchOff + exchBytes) {
    unsigned short* exch = (unsigned short*)((char*)d_ws + exchOff);
    const size_t lds = (size_t)(2 * 128 * SB) * sizeof(unsigned short);      // 266,240
    ctrnn_recur_fast<<<16, 256, lds, stream>>>(hidden, W_h, b_h, out, cnt, exch);
  } else {
    const size_t lds = (size_t)(2 * 128 * SB + 2 * 16 * SB) * sizeof(unsigned short);
    ctrnn_recur_staged<<<16, 256, lds, stream>>>(hidden, W_h, b_h, out, cnt);
  }
}
